// AE2TK_15925738734149
// MI455X (gfx1250) — matrix operand order repaired and checked by reading, not compiled, not run on hardware
//
#include <hip/hip_runtime.h>
#include <cstdint>
#include <cstddef>

// ---------------------------------------------------------------------------
// MoE block for MI455X (gfx1250): all GEMMs via V_WMMA_F32_16X16X32_F16.
// wave32, 256-thread blocks (8 waves), each wave computes a 32x32 C tile.
// ---------------------------------------------------------------------------

typedef __attribute__((ext_vector_type(16))) _Float16 v16h;
typedef __attribute__((ext_vector_type(8)))  _Float16 v8h;
typedef __attribute__((ext_vector_type(8)))  float    v8f;

static constexpr int  Bb   = 8;
static constexpr int  Ss   = 2048;
static constexpr int  Hh   = 1024;
static constexpr int  Ntok = Bb * Ss;          // 16384 rows in every GEMM
static constexpr float KEEP_P = 0.8f;
// floor(0.8 * 2^32)
static constexpr uint32_t KEEP_THRESH = 0xCCCCCCCCu;

// Deterministic per-element Bernoulli(KEEP) via splitmix64 hash.
__device__ __forceinline__ bool drop_keep(uint32_t seed, uint64_t idx) {
  uint64_t z = ((uint64_t)seed << 32) ^ (idx + 0x9E3779B97F4A7C15ull);
  z = (z ^ (z >> 30)) * 0xBF58476D1CE4E5B9ull;
  z = (z ^ (z >> 27)) * 0x94D049BB133111EBull;
  z ^= z >> 31;
  return (uint32_t)z < KEEP_THRESH;
}

// Build a v16h fragment from two 16-byte contiguous chunks.
__device__ __forceinline__ v16h make_frag(const _Float16* lo, const _Float16* hi) {
  v8h l = *(const v8h*)lo;
  v8h h = *(const v8h*)hi;
  v16h r;
#pragma unroll
  for (int i = 0; i < 8; ++i) { r[i] = l[i]; r[i + 8] = h[i]; }
  return r;
}

// ---------------------------------------------------------------------------
// hs = x + 0.002*noise  (kept in f32 for final combine/recon, f16 for GEMM A)
// ---------------------------------------------------------------------------
__global__ void prep_kernel(const float* __restrict__ x, const float* __restrict__ noise,
                            float* __restrict__ hs, _Float16* __restrict__ hsh, long total) {
  long i = (long)blockIdx.x * blockDim.x + threadIdx.x;
  if (i < total) {
    float v = x[i] + 0.002f * noise[i];
    hs[i]  = v;
    hsh[i] = (_Float16)v;
  }
}

// ---------------------------------------------------------------------------
// Pack row-major f32 weights W[K x Nout] into WMMA B-fragment order (f16):
//   tile t = nt*(K/32)+kt ; within tile, lane l holds 16 contiguous values
//   value v of lane l = W[kt*32 + kr + v, nt*16 + n], n=l&15, kr=16*(l>=16)
// ---------------------------------------------------------------------------
__global__ void pack_w(const float* __restrict__ W, _Float16* __restrict__ Wp,
                       int K, int Nout) {
  int idx = blockIdx.x * 256 + threadIdx.x;
  if (idx >= K * Nout) return;
  int v  = idx & 15;
  int l  = (idx >> 4) & 31;
  int t  = idx >> 9;
  int K32 = K >> 5;
  int kt = t % K32;
  int nt = t / K32;
  int n  = l & 15;
  int kr = (l < 16) ? 0 : 16;
  int kk = kt * 32 + kr + v;
  int nn = nt * 16 + n;
  Wp[idx] = (_Float16)W[(size_t)kk * Nout + nn];
}

// ---------------------------------------------------------------------------
// WMMA GEMM: C[M x Nout] = A[M x K] @ Wpacked + bias, fused epilogue.
//   EPI 0: f16 out = dropout(tanh(acc+b))   (down projections)
//   EPI 1: f16 out = tanh(acc+b)            (sub up projections)
//   EPI 2: f32 out = acc+b                  (final up projection)
// Grid: (M/64, Nout/128), block 256 threads (8 waves, 2x4), wave tile 32x32.
// ---------------------------------------------------------------------------
template <int EPI>
__global__ __launch_bounds__(256) void gemm_wmma(
    const _Float16* __restrict__ A, const _Float16* __restrict__ Wp,
    const float* __restrict__ bias, _Float16* __restrict__ outh,
    float* __restrict__ outf, int K, int Nout, uint32_t seed) {
  const int lane  = threadIdx.x & 31;
  const int wave  = threadIdx.x >> 5;
  const int waveM = wave & 1;
  const int waveN = wave >> 1;
  const int mln   = lane & 15;
  const int mhalf = lane >> 4;
  const int rowBase = blockIdx.x * 64 + waveM * 32;
  const int colBase = blockIdx.y * 128 + waveN * 32;
  const int K32 = K >> 5;

  const v8f zv = {0.f, 0.f, 0.f, 0.f, 0.f, 0.f, 0.f, 0.f};
  v8f acc[2][2];
  acc[0][0] = zv; acc[0][1] = zv; acc[1][0] = zv; acc[1][1] = zv;

  // A: row-major f16; lane holds row rowBase+mln (and +16), K chunk per layout.
  const _Float16* arow0 = A + (size_t)(rowBase + mln) * K + mhalf * 16;
  const _Float16* arow1 = arow0 + (size_t)16 * K;
  // B: packed; each k-step advances 512 f16 (one 32x16 tile).
  const _Float16* bp0 = Wp + ((size_t)(colBase >> 4) * K32) * 512 + (size_t)lane * 16;
  const _Float16* bp1 = bp0 + (size_t)K32 * 512;

  for (int kt = 0; kt < K32; ++kt) {
    const int k = kt << 5;
    v16h a0 = make_frag(arow0 + k, arow0 + k + 8);
    v16h a1 = make_frag(arow1 + k, arow1 + k + 8);
    v16h b0 = make_frag(bp0 + (size_t)kt * 512, bp0 + (size_t)kt * 512 + 8);
    v16h b1 = make_frag(bp1 + (size_t)kt * 512, bp1 + (size_t)kt * 512 + 8);
    acc[0][0] = __builtin_amdgcn_wmma_f32_16x16x32_f16(false, a0, false, b0, (short)0, acc[0][0], false, false);
    acc[0][1] = __builtin_amdgcn_wmma_f32_16x16x32_f16(false, a0, false, b1, (short)0, acc[0][1], false, false);
    acc[1][0] = __builtin_amdgcn_wmma_f32_16x16x32_f16(false, a1, false, b0, (short)0, acc[1][0], false, false);
    acc[1][1] = __builtin_amdgcn_wmma_f32_16x16x32_f16(false, a1, false, b1, (short)0, acc[1][1], false, false);
  }

  // D layout: VGPR r -> rows (mhalf*8 + r), lane -> column (mln).
#pragma unroll
  for (int mt = 0; mt < 2; ++mt) {
#pragma unroll
    for (int nt = 0; nt < 2; ++nt) {
      const int ncol = colBase + nt * 16 + mln;
      const int mrow = rowBase + mt * 16 + mhalf * 8;
      const float bv = bias[ncol];
#pragma unroll
      for (int r = 0; r < 8; ++r) {
        const int m = mrow + r;
        float v = acc[mt][nt][r] + bv;
        const size_t oidx = (size_t)m * Nout + ncol;
        if (EPI == 0) {
          v = tanhf(v);
          v = drop_keep(seed, oidx) ? v * (1.0f / KEEP_P) : 0.0f;
          outh[oidx] = (_Float16)v;
        } else if (EPI == 1) {
          outh[oidx] = (_Float16)tanhf(v);
        } else {
          outf[oidx] = v;
        }
      }
    }
  }
}

// ---------------------------------------------------------------------------
// hmix = onehot(sub_choice[e]) . {s0, s1, h}   (hard select)
// ---------------------------------------------------------------------------
__global__ void mix_kernel(const _Float16* __restrict__ s0, const _Float16* __restrict__ s1,
                           const _Float16* __restrict__ h, const int* __restrict__ subc,
                           int e, _Float16* __restrict__ hm, long n) {
  long i = (long)blockIdx.x * 256 + threadIdx.x;
  if (i >= n) return;
  int c = subc[e];
  hm[i] = (c == 0) ? s0[i] : ((c == 1) ? s1[i] : h[i]);
}

__global__ void zero_acc(float* acc) {
  if (threadIdx.x == 0 && blockIdx.x == 0) { acc[0] = 0.f; acc[1] = 0.f; }
}

// ---------------------------------------------------------------------------
// opt = route-select(h0,h1,hs) ; accumulate recon sums
// ---------------------------------------------------------------------------
__global__ void final_kernel(const float* __restrict__ hs, const float* __restrict__ h0,
                             const float* __restrict__ h1, const int* __restrict__ route,
                             float* __restrict__ out, float* __restrict__ acc, long total) {
  __shared__ float r0[256];
  __shared__ float r1[256];
  long i = (long)blockIdx.x * 256 + threadIdx.x;
  float d0 = 0.f, d1 = 0.f;
  if (i < total) {
    int s = (int)((i / Hh) % Ss);
    int r = route[s];
    float a = hs[i], b = h0[i], c = h1[i];
    out[i] = (r == 0) ? b : ((r == 1) ? c : a);
    float e0 = a - b, e1 = a - c;
    d0 = e0 * e0;
    d1 = e1 * e1;
  }
  r0[threadIdx.x] = d0;
  r1[threadIdx.x] = d1;
  __syncthreads();
  for (int sft = 128; sft > 0; sft >>= 1) {
    if ((int)threadIdx.x < sft) {
      r0[threadIdx.x] += r0[threadIdx.x + sft];
      r1[threadIdx.x] += r1[threadIdx.x + sft];
    }
    __syncthreads();
  }
  if (threadIdx.x == 0) {
    atomicAdd(&acc[0], r0[0]);
    atomicAdd(&acc[1], r1[0]);
  }
}

__global__ void write_recon(const float* __restrict__ acc, float* __restrict__ out, long total) {
  if (threadIdx.x == 0 && blockIdx.x == 0)
    out[0] = 0.5f * (acc[0] + acc[1]) / (float)total;
}

// ---------------------------------------------------------------------------
// Host orchestration
// ---------------------------------------------------------------------------
struct ExpertPtrs {
  const float *dw, *db, *s0dw, *s0db, *s0uw, *s0ub, *s1dw, *s1db, *s1uw, *s1ub, *uw, *ub;
};

extern "C" void kernel_launch(void* const* d_in, const int* in_sizes, int n_in,
                              void* d_out, int out_size, void* d_ws, size_t ws_size,
                              hipStream_t stream) {
  (void)in_sizes; (void)n_in; (void)out_size; (void)ws_size;

  const float* x     = (const float*)d_in[0];
  const float* noise = (const float*)d_in[1];
  const int*   route = (const int*)d_in[2];
  const int*   subc  = (const int*)d_in[3];

  ExpertPtrs ep[2];
  for (int e = 0; e < 2; ++e) {
    int b = 4 + e * 12;
    ep[e].dw   = (const float*)d_in[b + 0];  ep[e].db   = (const float*)d_in[b + 1];
    ep[e].s0dw = (const float*)d_in[b + 2];  ep[e].s0db = (const float*)d_in[b + 3];
    ep[e].s0uw = (const float*)d_in[b + 4];  ep[e].s0ub = (const float*)d_in[b + 5];
    ep[e].s1dw = (const float*)d_in[b + 6];  ep[e].s1db = (const float*)d_in[b + 7];
    ep[e].s1uw = (const float*)d_in[b + 8];  ep[e].s1ub = (const float*)d_in[b + 9];
    ep[e].uw   = (const float*)d_in[b + 10]; ep[e].ub   = (const float*)d_in[b + 11];
  }
  const int dims[2] = {512, 256};

  // ---- workspace layout (deterministic, 256B-aligned) ----
  char* ws = (char*)d_ws;
  size_t off = 0;
  auto alloc = [&](size_t bytes) -> char* {
    char* p = ws + off;
    off = (off + bytes + 255) & ~(size_t)255;
    return p;
  };
  const long NH = (long)Ntok * Hh;

  float*    hs   = (float*)   alloc((size_t)NH * 4);
  _Float16* hsh  = (_Float16*)alloc((size_t)NH * 2);
  float*    h0o  = (float*)   alloc((size_t)NH * 4);
  float*    h1o  = (float*)   alloc((size_t)NH * 4);
  float*    acc  = (float*)   alloc(256);

  // packed weights: [e][0..5] = dw, s0dw, s0uw, s1dw, s1uw, uw
  _Float16* pk[2][6];
  for (int e = 0; e < 2; ++e) {
    int d = dims[e];
    pk[e][0] = (_Float16*)alloc((size_t)Hh * d * 2);
    pk[e][1] = (_Float16*)alloc((size_t)d * (d / 2) * 2);
    pk[e][2] = (_Float16*)alloc((size_t)(d / 2) * d * 2);
    pk[e][3] = (_Float16*)alloc((size_t)d * (d / 2) * 2);
    pk[e][4] = (_Float16*)alloc((size_t)(d / 2) * d * 2);
    pk[e][5] = (_Float16*)alloc((size_t)d * Hh * 2);
  }

  // expert scratch, sized for d=512, reused for both experts
  _Float16* hbuf = (_Float16*)alloc((size_t)Ntok * 512 * 2);
  _Float16* t0b  = (_Float16*)alloc((size_t)Ntok * 256 * 2);
  _Float16* t1b  = (_Float16*)alloc((size_t)Ntok * 256 * 2);
  _Float16* s0b  = (_Float16*)alloc((size_t)Ntok * 512 * 2);
  _Float16* s1b  = (_Float16*)alloc((size_t)Ntok * 512 * 2);
  _Float16* hmix = (_Float16*)alloc((size_t)Ntok * 512 * 2);

  // ---- prep: hs = x + 0.002*noise ----
  prep_kernel<<<dim3((unsigned)((NH + 255) / 256)), dim3(256), 0, stream>>>(x, noise, hs, hsh, NH);

  // ---- pack all weights (tiny; lives in L2 for the whole launch) ----
  auto pack = [&](const float* W, _Float16* Wp, int K, int Nout) {
    int total = K * Nout;
    pack_w<<<dim3((total + 255) / 256), dim3(256), 0, stream>>>(W, Wp, K, Nout);
  };
  for (int e = 0; e < 2; ++e) {
    int d = dims[e];
    pack(ep[e].dw,   pk[e][0], Hh,    d);
    pack(ep[e].s0dw, pk[e][1], d,     d / 2);
    pack(ep[e].s0uw, pk[e][2], d / 2, d);
    pack(ep[e].s1dw, pk[e][3], d,     d / 2);
    pack(ep[e].s1uw, pk[e][4], d / 2, d);
    pack(ep[e].uw,   pk[e][5], d,     Hh);
  }

  // ---- expert chains ----
  auto run_expert = [&](int e, float* outF) {
    const int d = dims[e];
    const uint32_t s = 0x9E370001u + (uint32_t)e * 0x85EBCA77u;
    dim3 blk(256);
    // h = dropout(tanh(hs @ dw + db))
    gemm_wmma<0><<<dim3(Ntok / 64, d / 128), blk, 0, stream>>>(
        hsh, pk[e][0], ep[e].db, hbuf, nullptr, Hh, d, s + 1);
    // t0 = dropout(tanh(h @ s0dw + s0db)); s0 = tanh(t0 @ s0uw + s0ub)
    gemm_wmma<0><<<dim3(Ntok / 64, (d / 2) / 128), blk, 0, stream>>>(
        hbuf, pk[e][1], ep[e].s0db, t0b, nullptr, d, d / 2, s + 2);
    gemm_wmma<1><<<dim3(Ntok / 64, d / 128), blk, 0, stream>>>(
        t0b, pk[e][2], ep[e].s0ub, s0b, nullptr, d / 2, d, 0);
    // t1 / s1
    gemm_wmma<0><<<dim3(Ntok / 64, (d / 2) / 128), blk, 0, stream>>>(
        hbuf, pk[e][3], ep[e].s1db, t1b, nullptr, d, d / 2, s + 3);
    gemm_wmma<1><<<dim3(Ntok / 64, d / 128), blk, 0, stream>>>(
        t1b, pk[e][4], ep[e].s1ub, s1b, nullptr, d / 2, d, 0);
    // hmix = onehot(sub_choice[e]) . {s0, s1, h}
    long nd = (long)Ntok * d;
    mix_kernel<<<dim3((unsigned)((nd + 255) / 256)), blk, 0, stream>>>(
        s0b, s1b, hbuf, subc, e, hmix, nd);
    // out_e = hmix @ uw + ub   (f32)
    gemm_wmma<2><<<dim3(Ntok / 64, Hh / 128), blk, 0, stream>>>(
        hmix, pk[e][5], ep[e].ub, nullptr, outF, d, Hh, 0);
  };
  run_expert(0, h0o);
  run_expert(1, h1o);

  // ---- final combine + recon ----
  float* out = (float*)d_out;
  zero_acc<<<1, 32, 0, stream>>>(acc);
  final_kernel<<<dim3((unsigned)((NH + 255) / 256)), dim3(256), 0, stream>>>(
      hs, h0o, h1o, route, out, acc, NH);
  write_recon<<<1, 32, 0, stream>>>(acc, out + NH, NH);
}
